// Attention_42468636623492
// MI455X (gfx1250) — matrix operand order repaired and checked by reading, not compiled, not run on hardware
//
#include <hip/hip_runtime.h>
#include <hip/hip_fp16.h>

#define SEQ 4096
#define DIM 1024
#define NH  16
#define DH  64

typedef __attribute__((ext_vector_type(16))) _Float16 v16h;
typedef __attribute__((ext_vector_type(8)))  _Float16 v8h;
typedef __attribute__((ext_vector_type(8)))  float    v8f;

union FragU { v16h v; v8h h[2]; };

// ---------------- CDNA5 async global->LDS path (guarded) ---------------------
#if defined(__has_builtin)
#if __has_builtin(__builtin_amdgcn_global_load_async_to_lds_b128)
#define HAVE_ASYNC_LDS 1
#endif
#endif

#ifdef HAVE_ASYNC_LDS
// builtin signature: (V4i addrspace(1)* src, V4i addrspace(3)* dst, Imm offset, Imm cpol)
typedef int v4i_b128 __attribute__((vector_size(4 * sizeof(int))));
typedef __attribute__((address_space(1))) v4i_b128* g_b128_ptr;
typedef __attribute__((address_space(3))) v4i_b128* l_b128_ptr;
#endif

// copy 16 bytes global -> LDS (async on CDNA5, sync fallback otherwise)
__device__ __forceinline__ void copy16_g2l(const _Float16* gsrc, _Float16* ldst) {
#ifdef HAVE_ASYNC_LDS
  __builtin_amdgcn_global_load_async_to_lds_b128((g_b128_ptr)gsrc, (l_b128_ptr)ldst, 0, 0);
#else
  *(v8h*)ldst = *(const v8h*)gsrc;
#endif
}

template <int N>
__device__ __forceinline__ void async_wait_le() {
#ifdef HAVE_ASYNC_LDS
#if __has_builtin(__builtin_amdgcn_s_wait_asynccnt)
  __builtin_amdgcn_s_wait_asynccnt(N);
#else
  asm volatile("s_wait_asynccnt %0" :: "i"(N) : "memory");
#endif
#endif
}

// A fragment (16x32 f16), source row-major [m][k], ld in elements.
// Lane L: m = L%16; halfwords = K {hf*8..hf*8+7, 16+hf*8..16+hf*8+7}, hf=L/16.
__device__ __forceinline__ v16h load_frag_a(const _Float16* base, int ld, int m0, int k0, int lane) {
  const int l16 = lane & 15, hf = lane >> 4;
  const _Float16* row = base + (size_t)(m0 + l16) * (size_t)ld + k0;
  FragU f;
  f.h[0] = *(const v8h*)(row + hf * 8);
  f.h[1] = *(const v8h*)(row + 16 + hf * 8);
  return f.v;
}

// B fragment (32x16 f16) where source is stored "transposed-B": row-major [n][k].
// Lane L: n = L%16; halfwords = K {hf*8..hf*8+7, 16+hf*8..16+hf*8+7}, hf=L/16.
__device__ __forceinline__ v16h load_frag_b(const _Float16* base, int ld, int n0, int k0, int lane) {
  const int l16 = lane & 15, hf = lane >> 4;
  const _Float16* row = base + (size_t)(n0 + l16) * (size_t)ld + k0;
  FragU f;
  f.h[0] = *(const v8h*)(row + hf * 8);
  f.h[1] = *(const v8h*)(row + 16 + hf * 8);
  return f.v;
}

__device__ __forceinline__ v8f wmma32(v16h a, v16h b, v8f c) {
  return __builtin_amdgcn_wmma_f32_16x16x32_f16(false, a, false, b, (short)0, c, false, false);
}

// ---------------- Kernel 1: row-wise weight normalization -> f16 -------------
__global__ __launch_bounds__(256) void wnorm_kernel(const float* __restrict__ w,
                                                    _Float16* __restrict__ out) {
  const int row = blockIdx.x;
  const float* wr = w + (size_t)row * DIM;
  float ss = 0.f;
  for (int i = threadIdx.x; i < DIM; i += 256) { float v = wr[i]; ss += v * v; }
  for (int m = 16; m; m >>= 1) ss += __shfl_xor(ss, m, 32);
  __shared__ float red[8];
  if ((threadIdx.x & 31) == 0) red[threadIdx.x >> 5] = ss;
  __syncthreads();
  float tot = 0.f;
#pragma unroll
  for (int i = 0; i < 8; ++i) tot += red[i];
  const float scale = 1.0f / fmaxf(sqrtf(tot), 1e-12f);
  for (int i = threadIdx.x; i < DIM; i += 256)
    out[(size_t)row * DIM + i] = (_Float16)(wr[i] * scale);
}

// ---------------- Kernel 2: f32 -> f16 convert -------------------------------
__global__ __launch_bounds__(256) void cvt_kernel(const float* __restrict__ x,
                                                  _Float16* __restrict__ xh, int n) {
  int i = blockIdx.x * 256 + threadIdx.x;
  if (i < n) xh[i] = (_Float16)x[i];
}

// ---------------- Kernel 3: WMMA GEMM  Y = A(f16,[M][K]) * W^T(f16,[N][K]) ---
// mode 0: Q  -> l2norm per head row, *0.125, store f16 [h][n][dh]
// mode 1: K  -> l2norm per head row,        store f16 [h][n][dh]
// mode 2: V  ->                      store f16 transposed [h][dh][n]
// mode 3: O  ->                      store f32 [n][DIM] (d_out)
__global__ __launch_bounds__(256) void proj_gemm_kernel(const _Float16* __restrict__ A,
                                                        const _Float16* __restrict__ W,
                                                        void* __restrict__ out, int mode) {
  const int lane = threadIdx.x & 31;
  const int wave = threadIdx.x >> 5;
  const int l16 = lane & 15, hf = lane >> 4;
  const int m0 = blockIdx.x * 128 + wave * 16;   // rows of A
  const int n0 = blockIdx.y * 64;                // output cols (head*64 for modes 0-2)

  v8f acc[4] = {};
  for (int k = 0; k < DIM; k += 32) {
    v16h a = load_frag_a(A, DIM, m0, k, lane);
#pragma unroll
    for (int t = 0; t < 4; ++t) {
      v16h b = load_frag_b(W, DIM, n0 + t * 16, k, lane);
      acc[t] = wmma32(a, b, acc[t]);
    }
  }

  if (mode <= 1) {
    // per-row l2 norm over this head's 64 outputs (C layout: row = r + 8*hf)
#pragma unroll
    for (int r = 0; r < 8; ++r) {
      float ss = acc[0][r] * acc[0][r] + acc[1][r] * acc[1][r] +
                 acc[2][r] * acc[2][r] + acc[3][r] * acc[3][r];
      for (int m = 1; m < 16; m <<= 1) ss += __shfl_xor(ss, m, 32);
      float sc = 1.0f / fmaxf(sqrtf(ss), 1e-5f);
      if (mode == 0) sc *= 0.125f;               // fold softmax 1/sqrt(dh) into Q
#pragma unroll
      for (int t = 0; t < 4; ++t) acc[t][r] *= sc;
    }
  }

  if (mode <= 1) {
    _Float16* dst = (_Float16*)out;              // [h][n][dh]
    const size_t hbase = (size_t)blockIdx.y * SEQ * DH;
#pragma unroll
    for (int t = 0; t < 4; ++t)
#pragma unroll
      for (int r = 0; r < 8; ++r) {
        int n = m0 + r + 8 * hf;
        int d = t * 16 + l16;
        dst[hbase + (size_t)n * DH + d] = (_Float16)acc[t][r];
      }
  } else if (mode == 2) {
    _Float16* dst = (_Float16*)out;              // [h][dh][n]  (transposed V)
    const size_t hbase = (size_t)blockIdx.y * DH * SEQ;
#pragma unroll
    for (int t = 0; t < 4; ++t)
#pragma unroll
      for (int r = 0; r < 8; ++r) {
        int n = m0 + r + 8 * hf;
        int d = t * 16 + l16;
        dst[hbase + (size_t)d * SEQ + n] = (_Float16)acc[t][r];
      }
  } else {
    float* dst = (float*)out;                    // [n][DIM] f32
#pragma unroll
    for (int t = 0; t < 4; ++t)
#pragma unroll
      for (int r = 0; r < 8; ++r)
        dst[(size_t)(m0 + r + 8 * hf) * DIM + n0 + t * 16 + l16] = acc[t][r];
  }
}

// issue one 64-key K/Vt tile copy (4 x b128 per thread, async on CDNA5)
__device__ __forceinline__ void tile_loads(const _Float16* __restrict__ Khh,
                                           const _Float16* __restrict__ Vth,
                                           int j0, _Float16* kb, _Float16* vb, int tid) {
  // K tile: 64x64 f16 contiguous (4096 f16 = 512 chunks of 8)
  copy16_g2l(Khh + (size_t)j0 * DH + tid * 8,         kb + tid * 8);
  copy16_g2l(Khh + (size_t)j0 * DH + (tid + 256) * 8, kb + (tid + 256) * 8);
  // Vt tile: rows d=0..63, 64 f16 each at stride SEQ
  const int drow = tid >> 2, seg = tid & 3;
  const _Float16* vs = Vth + (size_t)drow * SEQ + j0 + seg * 16;
  copy16_g2l(vs,     vb + drow * 64 + seg * 16);
  copy16_g2l(vs + 8, vb + drow * 64 + seg * 16 + 8);
}

// ---------------- Kernel 4: flash attention (one head x 128 q-rows / block) --
// Double-buffered K/Vt tiles via async global->LDS; ASYNCcnt-paced pipeline.
__global__ __launch_bounds__(256) void attn_kernel(const _Float16* __restrict__ Qh,
                                                   const _Float16* __restrict__ Kh,
                                                   const _Float16* __restrict__ Vt,
                                                   const unsigned char* __restrict__ mask,
                                                   _Float16* __restrict__ Oh) {
  __shared__ __align__(16) _Float16 k_lds[2][64 * 64];   // [j][d] tiles
  __shared__ __align__(16) _Float16 v_lds[2][64 * 64];   // [d][j] tiles (from Vt)
  __shared__ __align__(16) _Float16 p_lds[8][16 * 32];   // per-wave P re-layout buffer

  const int lane = threadIdx.x & 31;
  const int wave = threadIdx.x >> 5;
  const int l16 = lane & 15, hf = lane >> 4;
  const int h = blockIdx.y;
  const int m0 = blockIdx.x * 128 + wave * 16;

  const _Float16* Qhh = Qh + (size_t)h * SEQ * DH;
  const _Float16* Khh = Kh + (size_t)h * SEQ * DH;
  const _Float16* Vth = Vt + (size_t)h * DH * SEQ;

  // Q fragments held in registers for the whole kernel (scale already folded in)
  const v16h aq0 = load_frag_a(Qhh, DH, m0, 0, lane);
  const v16h aq1 = load_frag_a(Qhh, DH, m0, 32, lane);

  v8f o[4] = {};
  float mrow[8], lrow[8];
#pragma unroll
  for (int r = 0; r < 8; ++r) { mrow[r] = -1e30f; lrow[r] = 0.f; }

  // prologue: prefetch tile 0
  tile_loads(Khh, Vth, 0, k_lds[0], v_lds[0], threadIdx.x);

  int cur = 0;
#pragma unroll 1
  for (int j0 = 0; j0 < SEQ; j0 += 64) {
    if (j0 + 64 < SEQ) {
      // issue next tile into the other buffer, then wait until only those
      // 4 newest per-thread async ops remain outstanding -> current tile ready
      tile_loads(Khh, Vth, j0 + 64, k_lds[cur ^ 1], v_lds[cur ^ 1], threadIdx.x);
      async_wait_le<4>();
    } else {
      async_wait_le<0>();
    }
    __syncthreads();
    const _Float16* kb = k_lds[cur];
    const _Float16* vb = v_lds[cur];

#pragma unroll
    for (int jj = 0; jj < 64; jj += 32) {
      // S = Q * K^T for 32 keys (two 16-col C tiles, K-dim = dh = 64)
      v8f s0 = {}, s1 = {};
      s0 = wmma32(aq0, load_frag_b(kb, 64, jj, 0, lane), s0);
      s0 = wmma32(aq1, load_frag_b(kb, 64, jj, 32, lane), s0);
      s1 = wmma32(aq0, load_frag_b(kb, 64, jj + 16, 0, lane), s1);
      s1 = wmma32(aq1, load_frag_b(kb, 64, jj + 16, 32, lane), s1);

      const bool ok0 = mask[j0 + jj + l16] != 0;
      const bool ok1 = mask[j0 + jj + 16 + l16] != 0;

      float corr[8];
#pragma unroll
      for (int r = 0; r < 8; ++r) {
        float v0 = ok0 ? s0[r] : -1e30f;
        float v1 = ok1 ? s1[r] : -1e30f;
        float mx = fmaxf(v0, v1);
        for (int m = 1; m < 16; m <<= 1) mx = fmaxf(mx, __shfl_xor(mx, m, 32));
        const float mnew = fmaxf(mrow[r], mx);
        const float c = __expf(mrow[r] - mnew);
        const float p0 = __expf(v0 - mnew);
        const float p1 = __expf(v1 - mnew);
        float ps = p0 + p1;
        for (int m = 1; m < 16; m <<= 1) ps += __shfl_xor(ps, m, 32);
        lrow[r] = lrow[r] * c + ps;
        mrow[r] = mnew;
        corr[r] = c;
        // stash P in A-source layout staging: row = r + 8*hf, cols l16 / l16+16
        const int prow = r + 8 * hf;
        p_lds[wave][prow * 32 + l16]      = (_Float16)p0;
        p_lds[wave][prow * 32 + 16 + l16] = (_Float16)p1;
      }
#pragma unroll
      for (int t = 0; t < 4; ++t)
#pragma unroll
        for (int r = 0; r < 8; ++r) o[t][r] *= corr[r];

      // O += P(16x32) * V(32x64): A from per-wave LDS stage, B from Vt tile
      const v16h ap = load_frag_a(p_lds[wave], 32, 0, 0, lane);
#pragma unroll
      for (int t = 0; t < 4; ++t)
        o[t] = wmma32(ap, load_frag_b(vb, 64, t * 16, jj, lane), o[t]);
    }
    __syncthreads();   // all waves done reading `cur` before it is refilled
    cur ^= 1;
  }

  // epilogue: divide by softmax denom, store f16 row-major [n][h*64+d]
#pragma unroll
  for (int t = 0; t < 4; ++t)
#pragma unroll
    for (int r = 0; r < 8; ++r) {
      const int n = m0 + r + 8 * hf;
      const float val = o[t][r] / lrow[r];
      Oh[(size_t)n * DIM + h * DH + t * 16 + l16] = (_Float16)val;
    }
}

// ---------------- launch ------------------------------------------------------
extern "C" void kernel_launch(void* const* d_in, const int* in_sizes, int n_in,
                              void* d_out, int out_size, void* d_ws, size_t ws_size,
                              hipStream_t stream) {
  const float* x = (const float*)d_in[0];
  const unsigned char* mask = (const unsigned char*)d_in[1];
  const float* w_q = (const float*)d_in[2];
  const float* w_k = (const float*)d_in[3];
  const float* w_v = (const float*)d_in[4];
  const float* w_o = (const float*)d_in[5];

  char* ws = (char*)d_ws;
  const size_t MB = 1u << 20;
  _Float16* xh  = (_Float16*)(ws);               // 8 MB: x in f16; reused as Oh later
  _Float16* Oh  = xh;                            // alias (xh dead after projections)
  _Float16* wnq = (_Float16*)(ws + 8 * MB);      // 4 x 2 MB normalized weights (f16)
  _Float16* wnk = wnq + (size_t)DIM * DIM;
  _Float16* wnv = wnk + (size_t)DIM * DIM;
  _Float16* wno = wnv + (size_t)DIM * DIM;
  _Float16* Qh  = (_Float16*)(ws + 16 * MB);     // [h][n][dh] 8 MB
  _Float16* Kh  = (_Float16*)(ws + 24 * MB);     // [h][n][dh] 8 MB
  _Float16* Vt  = (_Float16*)(ws + 32 * MB);     // [h][dh][n] 8 MB  (total 40 MB)

  // 1) normalize weight rows -> f16
  wnorm_kernel<<<DIM, 256, 0, stream>>>(w_q, wnq);
  wnorm_kernel<<<DIM, 256, 0, stream>>>(w_k, wnk);
  wnorm_kernel<<<DIM, 256, 0, stream>>>(w_v, wnv);
  wnorm_kernel<<<DIM, 256, 0, stream>>>(w_o, wno);

  // 2) x -> f16
  const int nel = SEQ * DIM;
  cvt_kernel<<<(nel + 255) / 256, 256, 0, stream>>>(x, xh, nel);

  // 3) projections (WMMA GEMMs)
  dim3 g(SEQ / 128, NH), b(256);
  proj_gemm_kernel<<<g, b, 0, stream>>>(xh, wnq, Qh, 0);
  proj_gemm_kernel<<<g, b, 0, stream>>>(xh, wnk, Kh, 1);
  proj_gemm_kernel<<<g, b, 0, stream>>>(xh, wnv, Vt, 2);

  // 4) flash attention -> Oh (aliases xh; xh no longer read)
  attn_kernel<<<dim3(SEQ / 128, NH), b, 0, stream>>>(Qh, Kh, Vt, mask, Oh);

  // 5) output projection -> d_out (f32)
  proj_gemm_kernel<<<dim3(SEQ / 128, DIM / 64), b, 0, stream>>>(Oh, wno, d_out, 3);
}
